// InteractionHeteroConv_19490561589643
// MI455X (gfx1250) — matrix operand order repaired and checked by reading, not compiled, not run on hardware
//
#include <hip/hip_runtime.h>
#include <hip/hip_bf16.h>

typedef __attribute__((ext_vector_type(16))) _Float16 v16h;
typedef __attribute__((ext_vector_type(8)))  _Float16 v8h;
typedef __attribute__((ext_vector_type(8)))  float    v8f;

// LDS layout (bytes). Strides chosen so every v8h/v16h LDS access is 16B-aligned
// and rows land on different bank groups.
#define LDA1 400   // halves stride for the 16x384 A tile   (800 B/row)
#define LDH  132   // float stride for the 16x128 H tile    (528 B/row)
#define LDA2 144   // halves stride for the 16x128 A2 tile  (288 B/row)
#define SM_A1   0
#define SM_H    12800          // 16*400*2
#define SM_A2   21248          // + 16*132*4
#define SM_RED  25856          // + 16*144*2
#define SM_TOTAL 28032         // + 544*4

// ---------------------------------------------------------------------------
// Small prep kernels
// ---------------------------------------------------------------------------
__global__ void wtrans_f32_to_f16(const float* __restrict__ src,  // [K][Nout] f32
                                  _Float16* __restrict__ dst,     // [Nout][K] f16
                                  int K, int Nout) {
  int idx = blockIdx.x * blockDim.x + threadIdx.x;
  if (idx < K * Nout) {
    int o = idx / K, i = idx - o * K;
    dst[idx] = (_Float16)src[i * Nout + o];
  }
}

__global__ void init_msg_kernel(float* __restrict__ p, long n) {
  long i = (long)blockIdx.x * blockDim.x + threadIdx.x;
  if (i < n) p[i] = ((i & 255) < 128) ? -__builtin_inff() : 0.0f;  // [max|-inf, sum|0]
}

// One thread per (edge, feature): float atomic max + float atomic add.
__global__ void scatter_sum_max_kernel(const float* __restrict__ e,
                                       const int* __restrict__ dst,
                                       float* __restrict__ msg, int nE) {
  long idx = (long)blockIdx.x * blockDim.x + threadIdx.x;
  if (idx >= (long)nE * 128) return;
  int ed = (int)(idx >> 7), f = (int)(idx & 127);
  float v = e[idx];
  float* base = msg + (long)dst[ed] * 256;
  __hip_atomic_fetch_max(base + f, v, __ATOMIC_RELAXED, __HIP_MEMORY_SCOPE_AGENT);
  __hip_atomic_fetch_add(base + 128 + f, v, __ATOMIC_RELAXED, __HIP_MEMORY_SCOPE_AGENT);
}

// ---------------------------------------------------------------------------
// Fused 2-layer MLP engine (16 rows per block, 8 waves, wave32)
// ---------------------------------------------------------------------------
__device__ __forceinline__ void ln_stats(const float* __restrict__ h,
                                         float* __restrict__ red, int t) {
  const int row = t >> 4, seg = t & 15;
  float s = 0.f, s2 = 0.f;
  const float* p = h + row * LDH + seg * 8;
#pragma unroll
  for (int i = 0; i < 8; ++i) { float v = p[i]; s += v; s2 += v * v; }
  red[row * 16 + seg] = s;
  red[256 + row * 16 + seg] = s2;
  __syncthreads();
  if (t < 16) {
    float S = 0.f, S2 = 0.f;
#pragma unroll
    for (int i = 0; i < 16; ++i) { S += red[t * 16 + i]; S2 += red[256 + t * 16 + i]; }
    float mu = S * (1.f / 128.f);
    float var = S2 * (1.f / 128.f) - mu * mu;
    red[512 + t] = mu;
    red[528 + t] = rsqrtf(var + 1e-5f);
  }
  __syncthreads();
}

// One 16x16 output tile, K-loop of v_wmma_f32_16x16x32_f16.
// A tile in LDS (row-major halves, stride LDAH); B from f16 weights stored [out][K].
// Contiguous k order, shared by A and B (not the ISA table order):
//   A lane(l): row = l%16, element e = K k0+16g+e   (g = l/16)
//   B lane(l): col = n0 + l%16, element e = K k0+16g+e  -> contiguous v16h load
template <int K, int LDAH>
__device__ __forceinline__ v8f wmma_tile(const _Float16* __restrict__ ldsA,
                                         const _Float16* __restrict__ Wt,
                                         int n0, int lane) {
  const int g = lane >> 4;
  const int l16 = lane & 15;
  v8f c = {};
  const _Float16* arow = ldsA + l16 * LDAH + 16 * g;
  const _Float16* brow = Wt + (n0 + l16) * K + 16 * g;
#pragma unroll
  for (int k0 = 0; k0 < K; k0 += 32) {
    v8h alo = *(const v8h*)(arow + k0);
    v8h ahi = *(const v8h*)(arow + k0 + 8);
    v16h a;
#pragma unroll
    for (int i = 0; i < 8; ++i) { a[i] = alo[i]; a[i + 8] = ahi[i]; }
    const v16h b = *(const v16h*)(brow + k0);
    c = __builtin_amdgcn_wmma_f32_16x16x32_f16(false, a, false, b, (short)0, c,
                                               false, false);
  }
  return c;
}

// Runs Linear(384->128)+LN+ReLU, Linear(128->128)+LN.  Leaves pre-LN layer-2
// activations in Hs and {mu,rstd} in red[512..543]; caller applies gamma/beta,
// ReLU and the residual on store.
__device__ __forceinline__ void mlp2_engine(
    unsigned char* smem,
    const _Float16* __restrict__ Wt1, const float* __restrict__ b1,
    const float* __restrict__ g1, const float* __restrict__ be1,
    const _Float16* __restrict__ Wt2, const float* __restrict__ b2) {
  _Float16* A1 = (_Float16*)(smem + SM_A1);
  float*    Hs = (float*)(smem + SM_H);
  _Float16* A2 = (_Float16*)(smem + SM_A2);
  float*    red = (float*)(smem + SM_RED);
  const int t = threadIdx.x;
  const int wave = t >> 5, lane = t & 31;
  const int g = lane >> 4, l16 = lane & 15;
  const int n0 = wave * 16;

  // ---- layer 1: [16x384] x [384x128] ----
  v8f c = wmma_tile<384, LDA1>(A1, Wt1, n0, lane);
  float bias = b1[n0 + l16];
#pragma unroll
  for (int r = 0; r < 8; ++r)              // C/D layout: elem r -> row r+8g, col n0+l16
    Hs[(r + 8 * g) * LDH + n0 + l16] = c[r] + bias;
  __syncthreads();
  ln_stats(Hs, red, t);
  for (int idx = t; idx < 2048; idx += 256) {   // LN + ReLU + f16 repack
    int row = idx >> 7, col = idx & 127;
    float v = (Hs[row * LDH + col] - red[512 + row]) * red[528 + row] * g1[col] +
              be1[col];
    A2[row * LDA2 + col] = (_Float16)fmaxf(v, 0.f);
  }
  __syncthreads();

  // ---- layer 2: [16x128] x [128x128] ----
  v8f c2 = wmma_tile<128, LDA2>(A2, Wt2, n0, lane);
  float bias2 = b2[n0 + l16];
#pragma unroll
  for (int r = 0; r < 8; ++r)
    Hs[(r + 8 * g) * LDH + n0 + l16] = c2[r] + bias2;
  __syncthreads();
  ln_stats(Hs, red, t);
}

// ---------------------------------------------------------------------------
// Node update: in = [x | clamp(max) | sum], out = x + mlp(in)
// ---------------------------------------------------------------------------
__global__ __launch_bounds__(256) void node_mlp_kernel(
    const float* __restrict__ x, const float* __restrict__ msg,
    const _Float16* __restrict__ Wt1, const float* __restrict__ b1,
    const float* __restrict__ g1, const float* __restrict__ be1,
    const _Float16* __restrict__ Wt2, const float* __restrict__ b2,
    const float* __restrict__ g2, const float* __restrict__ be2,
    float* __restrict__ out, int nN) {
  __shared__ __align__(32) unsigned char smem[SM_TOTAL];
  _Float16* A1 = (_Float16*)(smem + SM_A1);
  const int t = threadIdx.x;
  const int nb = blockIdx.x * 16;
  for (int idx = t; idx < 16 * 384; idx += 256) {
    int row = idx / 384, k = idx - row * 384;
    int node = nb + row;
    float v = 0.f;
    if (node < nN) {
      if (k < 128) {
        v = x[(long)node * 128 + k];
      } else {
        v = msg[(long)node * 256 + (k - 128)];
        if (k < 256 && v == -__builtin_inff()) v = 0.f;  // empty segment -> 0
      }
    }
    A1[row * LDA1 + k] = (_Float16)v;
  }
  __syncthreads();
  mlp2_engine(smem, Wt1, b1, g1, be1, Wt2, b2);
  float* red = (float*)(smem + SM_RED);
  float* Hs = (float*)(smem + SM_H);
  for (int idx = t; idx < 2048; idx += 256) {
    int row = idx >> 7, col = idx & 127;
    int node = nb + row;
    if (node >= nN) continue;
    float v = (Hs[row * LDH + col] - red[512 + row]) * red[528 + row] * g2[col] +
              be2[col];
    out[(long)node * 128 + col] = x[(long)node * 128 + col] + fmaxf(v, 0.f);
  }
}

// ---------------------------------------------------------------------------
// Edge update: in = [x_src[ei0] | x_dst[ei1] | e], out = e + mlp(in)
// ---------------------------------------------------------------------------
__global__ __launch_bounds__(256) void edge_mlp_kernel(
    const float* __restrict__ xsrc, const float* __restrict__ xdst,
    const int* __restrict__ ei, const float* __restrict__ ef,
    const _Float16* __restrict__ Wt1, const float* __restrict__ b1,
    const float* __restrict__ g1, const float* __restrict__ be1,
    const _Float16* __restrict__ Wt2, const float* __restrict__ b2,
    const float* __restrict__ g2, const float* __restrict__ be2,
    float* __restrict__ out, int nE) {
  __shared__ __align__(32) unsigned char smem[SM_TOTAL];
  _Float16* A1 = (_Float16*)(smem + SM_A1);
  const int t = threadIdx.x;
  const int eb = blockIdx.x * 16;
  for (int idx = t; idx < 16 * 384; idx += 256) {
    int row = idx / 384, k = idx - row * 384;
    int e = eb + row;
    float v = 0.f;
    if (e < nE) {
      if (k < 128)      v = xsrc[(long)ei[e] * 128 + k];
      else if (k < 256) v = xdst[(long)ei[nE + e] * 128 + (k - 128)];
      else              v = ef[(long)e * 128 + (k - 256)];
    }
    A1[row * LDA1 + k] = (_Float16)v;
  }
  __syncthreads();
  mlp2_engine(smem, Wt1, b1, g1, be1, Wt2, b2);
  float* red = (float*)(smem + SM_RED);
  float* Hs = (float*)(smem + SM_H);
  for (int idx = t; idx < 2048; idx += 256) {
    int row = idx >> 7, col = idx & 127;
    int e = eb + row;
    if (e >= nE) continue;
    float v = (Hs[row * LDH + col] - red[512 + row]) * red[528 + row] * g2[col] +
              be2[col];
    out[(long)e * 128 + col] = ef[(long)e * 128 + col] + fmaxf(v, 0.f);
  }
}

// ---------------------------------------------------------------------------
// Host launch
// ---------------------------------------------------------------------------
extern "C" void kernel_launch(void* const* d_in, const int* in_sizes, int n_in,
                              void* d_out, int out_size, void* d_ws, size_t ws_size,
                              hipStream_t stream) {
  const int H = 128;
  const int N = in_sizes[0] / H;
  const int E = in_sizes[2] / H;

  const float* x_pixel = (const float*)d_in[0];
  const float* x_strip = (const float*)d_in[1];
  const float* e_in[4] = {(const float*)d_in[2], (const float*)d_in[3],
                          (const float*)d_in[4], (const float*)d_in[5]};
  const int* ei[4] = {(const int*)d_in[6], (const int*)d_in[7],
                      (const int*)d_in[8], (const int*)d_in[9]};
  // params flattened in dict order: node.pixel, node.strip, edge.{pp,ps,sp,ss};
  // each MLP = (W1,b1,g1,beta1, W2,b2,g2,beta2)
  const float* prm[6][8];
  for (int m = 0; m < 6; ++m)
    for (int j = 0; j < 8; ++j) prm[m][j] = (const float*)d_in[10 + m * 8 + j];

  // --- f16 transposed weights in workspace (~768 KB; stays in L2) ---
  _Float16* wbase = (_Float16*)d_ws;
  for (int m = 0; m < 6; ++m) {
    _Float16* Wt1 = wbase + (long)m * 65536;
    _Float16* Wt2 = Wt1 + 49152;
    wtrans_f32_to_f16<<<(384 * 128 + 255) / 256, 256, 0, stream>>>(prm[m][0], Wt1, 384, 128);
    wtrans_f32_to_f16<<<(128 * 128 + 255) / 256, 256, 0, stream>>>(prm[m][4], Wt2, 128, 128);
  }

  float* out = (float*)d_out;
  const long NH = (long)N * H, EH = (long)E * H;
  float* xp = out;
  float* xs = out + NH;
  float* eout[4] = {out + 2 * NH, out + 2 * NH + EH, out + 2 * NH + 2 * EH,
                    out + 2 * NH + 3 * EH};
  // msg scratch lives in the edge-output region (2*N*256 <= 4*E*128 floats);
  // node kernels consume it before edge kernels overwrite it.
  float* msgP = out + 2 * NH;
  float* msgS = msgP + (long)N * 256;

  const long msgElems = 2L * N * 256;
  init_msg_kernel<<<(int)((msgElems + 255) / 256), 256, 0, stream>>>(msgP, msgElems);

  const int sblocks = (int)(((long)E * H + 255) / 256);
  scatter_sum_max_kernel<<<sblocks, 256, 0, stream>>>(e_in[0], ei[0] + E, msgP, E); // pp -> pixel
  scatter_sum_max_kernel<<<sblocks, 256, 0, stream>>>(e_in[2], ei[2] + E, msgP, E); // sp -> pixel
  scatter_sum_max_kernel<<<sblocks, 256, 0, stream>>>(e_in[1], ei[1] + E, msgS, E); // ps -> strip
  scatter_sum_max_kernel<<<sblocks, 256, 0, stream>>>(e_in[3], ei[3] + E, msgS, E); // ss -> strip

  const int nblocks = (N + 15) / 16;
  node_mlp_kernel<<<nblocks, 256, 0, stream>>>(
      x_pixel, msgP, wbase + 0 * 65536, prm[0][1], prm[0][2], prm[0][3],
      wbase + 0 * 65536 + 49152, prm[0][5], prm[0][6], prm[0][7], xp, N);
  node_mlp_kernel<<<nblocks, 256, 0, stream>>>(
      x_strip, msgS, wbase + 1 * 65536, prm[1][1], prm[1][2], prm[1][3],
      wbase + 1 * 65536 + 49152, prm[1][5], prm[1][6], prm[1][7], xs, N);

  const int eblocks = (E + 15) / 16;
  const float* srcs[4] = {xp, xp, xs, xs};
  const float* dsts[4] = {xp, xs, xp, xs};
  for (int q = 0; q < 4; ++q) {
    const int m = 2 + q;
    edge_mlp_kernel<<<eblocks, 256, 0, stream>>>(
        srcs[q], dsts[q], ei[q], e_in[q],
        wbase + (long)m * 65536, prm[m][1], prm[m][2], prm[m][3],
        wbase + (long)m * 65536 + 49152, prm[m][5], prm[m][6], prm[m][7],
        eout[q], E);
  }
}
